// SparseLinear_34076270526541
// MI455X (gfx1250) — matrix operand order repaired and checked by reading, not compiled, not run on hardware
//
#include <hip/hip_runtime.h>
#include <hip/hip_bf16.h>
#include <stdint.h>

// ---------------- problem constants (from reference) ----------------
#define IN_F   4096
#define OUT_F  4096
#define MROWS  512                      // 2 * 256 tokens, flattened
#define TOTAL  (IN_F * OUT_F)           // 16,777,216
#define NUM_DROP 251658                 // int(NUM_NONZERO * 0.15) per reference

typedef __bf16 bf16_t;
typedef __attribute__((ext_vector_type(16))) bf16_t v16bf;
typedef __attribute__((ext_vector_type(8)))  bf16_t v8bf;
typedef __attribute__((ext_vector_type(8)))  float  v8f;
typedef __attribute__((ext_vector_type(4)))  uint32_t v4u;
typedef __attribute__((ext_vector_type(8)))  int      v8i;
typedef __attribute__((ext_vector_type(4)))  int      v4i;

// ---------------- workspace layout (bytes) ----------------
#define WS_HISTW_C   0                  // 32768 u32  (128 KB)
#define WS_HISTG_C   131072             // 32768 u32
#define WS_HISTW_F   262144             // 256 u32
#define WS_HISTG_F   263168             // 256 u32
#define WS_SCAL      264192             // 8 u32: [0]=binW [1]=aboveW [2]=binG [3]=aboveG [4]=dropT [5]=growT
#define WS_XB        (1u<<20)           // 512*4096 bf16 = 4 MB
#define WS_WB        (8u<<20)           // 4096*4096 bf16 = 32 MB
#define WS_NEEDED    (40u<<20)
#define WS_ZERO_U32  66056

__device__ __forceinline__ uint32_t abs_bits(float f) {
    union { float f; uint32_t u; } c; c.f = f;
    return c.u & 0x7FFFFFFFu;           // monotonic key for |f|
}
__device__ __forceinline__ unsigned short f2bf_rne(float f) {
    union { float f; uint32_t u; } c; c.f = f;
    uint32_t r = c.u + 0x7FFFu + ((c.u >> 16) & 1u);   // round-to-nearest-even
    return (unsigned short)(r >> 16);
}

// ---------------- phase 0: clear histograms + scalars ----------------
__global__ void k_zero(uint32_t* ws) {
    uint32_t i = blockIdx.x * blockDim.x + threadIdx.x;
    if (i < WS_ZERO_U32) ws[i] = 0u;
}

// ---------------- phase 1: coarse histograms (top 15 abs bits) ----------------
__global__ void k_hist_coarse(const float* __restrict__ w, const float* __restrict__ g,
                              const unsigned char* __restrict__ mask,
                              uint32_t* __restrict__ histW, uint32_t* __restrict__ histG) {
    uint32_t stride = gridDim.x * blockDim.x;
    for (uint32_t i = blockIdx.x * blockDim.x + threadIdx.x; i < TOTAL; i += stride) {
        atomicAdd(&histW[abs_bits(w[i]) >> 16], 1u);
        if (!mask[i]) atomicAdd(&histG[abs_bits(g[i]) >> 16], 1u);
    }
}

// ---------------- phase 2: scan coarse from the top ----------------
__global__ void k_scan_coarse(const uint32_t* __restrict__ histW,
                              const uint32_t* __restrict__ histG,
                              uint32_t* __restrict__ scal) {
    uint64_t cum = 0; uint32_t bin = 0, above = 0; bool found = false;
    for (int i = 32767; i >= 0; --i) {
        uint64_t h = histW[i];
        if (!found && cum + h >= (uint64_t)NUM_DROP) { bin = (uint32_t)i; above = (uint32_t)cum; found = true; }
        cum += h;
    }
    scal[0] = bin; scal[1] = above;
    cum = 0; bin = 0; above = 0; found = false;
    for (int i = 32767; i >= 0; --i) {
        uint64_t h = histG[i];
        if (!found && cum + h >= (uint64_t)NUM_DROP) { bin = (uint32_t)i; above = (uint32_t)cum; found = true; }
        cum += h;
    }
    scal[2] = bin; scal[3] = above;
}

// ---------------- phase 3: fine histograms (next 8 bits inside threshold bin) ----------------
__global__ void k_hist_fine(const float* __restrict__ w, const float* __restrict__ g,
                            const unsigned char* __restrict__ mask,
                            const uint32_t* __restrict__ scal,
                            uint32_t* __restrict__ histWf, uint32_t* __restrict__ histGf) {
    uint32_t bw = scal[0], bg = scal[2];
    uint32_t stride = gridDim.x * blockDim.x;
    for (uint32_t i = blockIdx.x * blockDim.x + threadIdx.x; i < TOTAL; i += stride) {
        uint32_t kw = abs_bits(w[i]);
        if ((kw >> 16) == bw) atomicAdd(&histWf[(kw >> 8) & 0xFFu], 1u);
        if (!mask[i]) {
            uint32_t kg = abs_bits(g[i]);
            if ((kg >> 16) == bg) atomicAdd(&histGf[(kg >> 8) & 0xFFu], 1u);
        }
    }
}

// ---------------- phase 4: final uint thresholds ----------------
__global__ void k_scan_fine(const uint32_t* __restrict__ histWf,
                            const uint32_t* __restrict__ histGf,
                            uint32_t* __restrict__ scal) {
    uint32_t needW = (uint32_t)NUM_DROP - scal[1];
    uint64_t cum = 0; uint32_t sub = 0; bool found = false;
    for (int i = 255; i >= 0; --i) {
        uint64_t h = histWf[i];
        if (!found && cum + h >= (uint64_t)needW) { sub = (uint32_t)i; found = true; }
        cum += h;
    }
    scal[4] = (scal[0] << 16) | (sub << 8);         // drop if abs_bits(w) >= dropT

    uint32_t needG = (uint32_t)NUM_DROP - scal[3];
    cum = 0; sub = 0; found = false;
    for (int i = 255; i >= 0; --i) {
        uint64_t h = histGf[i];
        if (!found && cum + h >= (uint64_t)needG) { sub = (uint32_t)i; found = true; }
        cum += h;
    }
    scal[5] = (scal[2] << 16) | (sub << 8);         // grow if !mask && abs_bits(g) >= growT
}

// ---------------- phase 5: build masked W in bf16 ----------------
__global__ void k_build_wb(const float* __restrict__ w, const float* __restrict__ g,
                           const unsigned char* __restrict__ mask,
                           const uint32_t* __restrict__ scal,
                           unsigned short* __restrict__ wb) {
    uint32_t dropT = scal[4], growT = scal[5];
    uint32_t stride = gridDim.x * blockDim.x;
    for (uint32_t i = blockIdx.x * blockDim.x + threadIdx.x; i < TOTAL; i += stride) {
        float wv   = w[i];
        bool drop  = abs_bits(wv) >= dropT;                    // new_mask = ones, minus top-|W|
        bool grow  = (!mask[i]) && (abs_bits(g[i]) >= growT);  // re-enable by |grad| among inactive
        float kept = (!drop || grow) ? wv : 0.0f;
        wb[i] = f2bf_rne(kept);
    }
}

// ---------------- phase 6: x -> bf16 ----------------
__global__ void k_convert_x(const float* __restrict__ x, unsigned short* __restrict__ xb) {
    uint32_t i = blockIdx.x * blockDim.x + threadIdx.x;
    if (i < (uint32_t)(MROWS * IN_F)) xb[i] = f2bf_rne(x[i]);
}

// ---------------- TDM: 2-D tile (64 rows x 64 bf16) global -> LDS ----------------
// D# per CDNA5 ISA ch.8: group0 = {count, lds_addr, global_addr, type=2},
// group1 = {wg_mask=0, data_size=1(2B), tensor_dim0/1=4096, tile_dim0/1=64, dim0_stride=4096}
// This toolchain's builtin takes 6 args (g0, g1, g2, g3, extra_group, cpol).
__device__ __forceinline__ void tdm_load_tile64x64(uint64_t gaddr, uint32_t lds_addr) {
    v4u g0;
    g0[0] = 1u;                                            // count=1, user descriptor
    g0[1] = lds_addr;                                      // LDS byte address
    g0[2] = (uint32_t)gaddr;                               // global_addr[31:0]
    g0[3] = ((uint32_t)(gaddr >> 32) & 0x1FFFFFFu) | (2u << 30);  // addr[56:32], type=2
    v8i g1;
    g1[0] = (int)(1u << 16);          // workgroup_mask=0, data_size=1 (2 bytes)
    g1[1] = (int)(0x1000u << 16);     // tensor_dim0 = 4096 (bits 79:48 low half)
    g1[2] = (int)(0x1000u << 16);     // tensor_dim0 hi=0, tensor_dim1 = 4096 (low16)
    g1[3] = (int)(64u << 16);         // tensor_dim1 hi=0, tile_dim0 = 64
    g1[4] = 64;                       // tile_dim1 = 64, tile_dim2 = 0
    g1[5] = 4096;                     // tensor_dim0_stride[31:0] = 4096 elements
    g1[6] = 0;                        // stride hi, tensor_dim1_stride lo
    g1[7] = 0;
    v4i z4 = {0, 0, 0, 0};            // groups 2/3 unused (2-D tile)
    v8i z8 = {0, 0, 0, 0, 0, 0, 0, 0};
    __builtin_amdgcn_tensor_load_to_lds(g0, g1, z4, z4, z8, 0);
}

// ---------------- phase 7: WMMA GEMM with TDM-staged B tiles ----------------
// Block = 8 waves sharing one 64-col N group; wave w owns M tile (blk&3)*8+w.
// K loop in 64-wide chunks: B chunk (64 rows x 64 k, 8 KB) DMA'd to LDS, double buffered.
#define KC 64
#define BUF_ELEMS (64 * KC)             // bf16 elements per buffer (8 KB)

__global__ void __launch_bounds__(256)
k_gemm_wmma(const bf16_t* __restrict__ xb, const unsigned short* __restrict__ wb_raw,
            float* __restrict__ out) {
    __shared__ unsigned short ldsB[2 * BUF_ELEMS];         // 16 KB, double buffered

    const int lane  = threadIdx.x & 31;
    const int wave  = threadIdx.x >> 5;
    const int ng    = blockIdx.x >> 2;                     // 64 N groups of 64 cols
    const int mt    = (blockIdx.x & 3) * 8 + wave;         // 32 M tiles of 16 rows
    const int row16 = lane & 15;
    const int hi    = lane >> 4;

    const bf16_t* arow = xb + (size_t)(mt * 16 + row16) * IN_F;
    const uint64_t wb_base = (uint64_t)(uintptr_t)wb_raw + (uint64_t)(ng * 64) * IN_F * 2u;
    const uint32_t lds_base = (uint32_t)(uintptr_t)(&ldsB[0]);

    v8f acc0 = {}, acc1 = {}, acc2 = {}, acc3 = {};

    const int nchunks = IN_F / KC;                          // 64
    if (wave == 0) tdm_load_tile64x64(wb_base, lds_base);   // prologue: chunk 0 -> buf 0

    for (int c = 0; c < nchunks; ++c) {
        if (wave == 0) __builtin_amdgcn_s_wait_tensorcnt(0);   // chunk c landed in LDS
        __syncthreads();                                        // all waves: buf[c&1] ready,
                                                                // and buf[(c+1)&1] fully consumed
        if (wave == 0 && c + 1 < nchunks)
            tdm_load_tile64x64(wb_base + (uint64_t)(c + 1) * KC * 2u,
                               lds_base + ((c + 1) & 1) * (BUF_ELEMS * 2u));

        const unsigned short* bbuf = &ldsB[(c & 1) * BUF_ELEMS];

        #pragma unroll
        for (int kk = 0; kk < KC; kk += 32) {
            const int k = c * KC + kk;
            // A: 16-bit A-matrix 16x32 (lanes 0-15: K=0..7,16..23; lanes 16-31: K=8..15,24..31)
            v8bf a_lo = *(const v8bf*)(arow + k + hi * 8);
            v8bf a_hi = *(const v8bf*)(arow + k + 16 + hi * 8);
            v16bf A = __builtin_shufflevector(a_lo, a_hi,
                                              0,1,2,3,4,5,6,7,8,9,10,11,12,13,14,15);
            // B from LDS: row (t*16+row16) of the 64x64 tile, K runs hi*8..+7 and 16+hi*8..+7
            #pragma unroll
            for (int t = 0; t < 4; ++t) {
                const unsigned short* bp = bbuf + (t * 16 + row16) * KC + kk + hi * 8;
                v8bf b0 = *(const v8bf*)(bp);
                v8bf b1 = *(const v8bf*)(bp + 16);
                v16bf B = __builtin_shufflevector(b0, b1,
                                                  0,1,2,3,4,5,6,7,8,9,10,11,12,13,14,15);
                if      (t == 0) acc0 = __builtin_amdgcn_wmma_f32_16x16x32_bf16(false, A, false, B, (short)0, acc0, false, false);
                else if (t == 1) acc1 = __builtin_amdgcn_wmma_f32_16x16x32_bf16(false, A, false, B, (short)0, acc1, false, false);
                else if (t == 2) acc2 = __builtin_amdgcn_wmma_f32_16x16x32_bf16(false, A, false, B, (short)0, acc2, false, false);
                else             acc3 = __builtin_amdgcn_wmma_f32_16x16x32_bf16(false, A, false, B, (short)0, acc3, false, false);
            }
        }
    }

    // C/D layout: VGPR r -> (M = r + hi*8, N = row16)
    const int rbase = mt * 16 + hi * 8;
    const int cbase = ng * 64 + row16;
    #pragma unroll
    for (int r = 0; r < 8; ++r) {
        out[(size_t)(rbase + r) * OUT_F + cbase +  0] = acc0[r];
        out[(size_t)(rbase + r) * OUT_F + cbase + 16] = acc1[r];
        out[(size_t)(rbase + r) * OUT_F + cbase + 32] = acc2[r];
        out[(size_t)(rbase + r) * OUT_F + cbase + 48] = acc3[r];
    }
}

// ---------------- host-side orchestration ----------------
extern "C" void kernel_launch(void* const* d_in, const int* in_sizes, int n_in,
                              void* d_out, int out_size, void* d_ws, size_t ws_size,
                              hipStream_t stream) {
    (void)in_sizes; (void)n_in; (void)out_size;
    if (ws_size < WS_NEEDED) return;

    const float*         x    = (const float*)d_in[0];          // [2,256,4096]
    const float*         w    = (const float*)d_in[1];          // [4096,4096]
    const float*         g    = (const float*)d_in[2];          // [4096,4096]
    const unsigned char* mask = (const unsigned char*)d_in[3];  // bool [4096,4096]
    float* out = (float*)d_out;

    uint8_t*  ws     = (uint8_t*)d_ws;
    uint32_t* histWc = (uint32_t*)(ws + WS_HISTW_C);
    uint32_t* histGc = (uint32_t*)(ws + WS_HISTG_C);
    uint32_t* histWf = (uint32_t*)(ws + WS_HISTW_F);
    uint32_t* histGf = (uint32_t*)(ws + WS_HISTG_F);
    uint32_t* scal   = (uint32_t*)(ws + WS_SCAL);
    unsigned short* xb = (unsigned short*)(ws + WS_XB);
    unsigned short* wb = (unsigned short*)(ws + WS_WB);

    k_zero<<<(WS_ZERO_U32 + 255) / 256, 256, 0, stream>>>((uint32_t*)ws);
    k_hist_coarse<<<2048, 256, 0, stream>>>(w, g, mask, histWc, histGc);
    k_scan_coarse<<<1, 1, 0, stream>>>(histWc, histGc, scal);
    k_hist_fine<<<2048, 256, 0, stream>>>(w, g, mask, scal, histWf, histGf);
    k_scan_fine<<<1, 1, 0, stream>>>(histWf, histGf, scal);
    k_build_wb<<<4096, 256, 0, stream>>>(w, g, mask, scal, wb);
    k_convert_x<<<(MROWS * IN_F + 255) / 256, 256, 0, stream>>>(x, xb);
    k_gemm_wmma<<<256, 256, 0, stream>>>((const bf16_t*)xb, wb, out);
}
